// WireframeDetector_73624329388553
// MI455X (gfx1250) — matrix operand order repaired and checked by reading, not compiled, not run on hardware
//
#include <hip/hip_runtime.h>
#include <hip/hip_bf16.h>
#include <math.h>

typedef __attribute__((ext_vector_type(16))) _Float16 v16h;
typedef __attribute__((ext_vector_type(8)))  _Float16 v8h;
typedef __attribute__((ext_vector_type(8)))  float    v8f;

#define HW       16384
#define IMG_W    128
#define N_LINES  10000
#define N_TOPK   300
#define C_FEAT   256
#define DIM_LOI  128
#define DIM_FC   1024

// ---------------------------------------------------------------------------
// f32 -> f16 conversion
// ---------------------------------------------------------------------------
__global__ void __launch_bounds__(256)
convert_f16_kernel(const float* __restrict__ in, _Float16* __restrict__ out, int n) {
    int i = blockIdx.x * blockDim.x + threadIdx.x;
    if (i < n) out[i] = (_Float16)in[i];
}

// ---------------------------------------------------------------------------
// jloc = softmax(ch5,ch6)[1] = sigmoid(c6-c5); NMS 3x3; joff = sigmoid(ch7,8)-0.5
// ---------------------------------------------------------------------------
__global__ void __launch_bounds__(256)
junction_prep_kernel(const float* __restrict__ out9,
                     float* __restrict__ jflat, float* __restrict__ joff) {
    int i = blockIdx.x * blockDim.x + threadIdx.x;
    if (i >= HW) return;
    int x = i & (IMG_W - 1);
    int y = i >> 7;
    const float* c5 = out9 + 5 * HW;
    const float* c6 = out9 + 6 * HW;
    auto jl = [&](int yy, int xx) {
        float a = c5[yy * IMG_W + xx], b = c6[yy * IMG_W + xx];
        return 1.0f / (1.0f + expf(a - b));
    };
    float v = jl(y, x);
    float m = v;
    #pragma unroll
    for (int dy = -1; dy <= 1; ++dy) {
        #pragma unroll
        for (int dx = -1; dx <= 1; ++dx) {
            int yy = y + dy, xx = x + dx;
            if (yy >= 0 && yy < IMG_W && xx >= 0 && xx < IMG_W)
                m = fmaxf(m, jl(yy, xx));
        }
    }
    jflat[i] = (v == m) ? v : 0.0f;
    joff[i]      = 1.0f / (1.0f + expf(-out9[7 * HW + i])) - 0.5f;
    joff[HW + i] = 1.0f / (1.0f + expf(-out9[8 * HW + i])) - 0.5f;
}

// ---------------------------------------------------------------------------
// iterative top-300 (single block, 1024 threads). Mutates jflat (rebuilt each call).
// Tie-break: lowest index wins (matches stable jax.lax.top_k).
// ---------------------------------------------------------------------------
__global__ void __launch_bounds__(1024)
topk_kernel(float* __restrict__ work, const float* __restrict__ joff,
            float* __restrict__ jscores_out, float* __restrict__ juncs_out,
            float* __restrict__ juncs_ws) {
    __shared__ float sval[1024];
    __shared__ int   sidx[1024];
    int tid = threadIdx.x;
    for (int t = 0; t < N_TOPK; ++t) {
        float bv = -INFINITY; int bi = 0x7fffffff;
        for (int i = tid; i < HW; i += 1024) {
            float v = work[i];
            if (v > bv) { bv = v; bi = i; }
        }
        sval[tid] = bv; sidx[tid] = bi;
        __syncthreads();
        for (int s = 512; s > 0; s >>= 1) {
            if (tid < s) {
                float ov = sval[tid + s]; int oi = sidx[tid + s];
                if (ov > sval[tid] || (ov == sval[tid] && oi < sidx[tid])) {
                    sval[tid] = ov; sidx[tid] = oi;
                }
            }
            __syncthreads();
        }
        if (tid == 0) {
            int idx = sidx[0];
            jscores_out[t] = sval[0];
            work[idx] = -INFINITY;
            float jx = (float)(idx & (IMG_W - 1)) + joff[idx] + 0.5f;
            float jy = (float)(idx >> 7) + joff[HW + idx] + 0.5f;
            juncs_out[2 * t] = jx; juncs_out[2 * t + 1] = jy;
            juncs_ws[2 * t]  = jx; juncs_ws[2 * t + 1]  = jy;
        }
        __syncthreads();
    }
}

// ---------------------------------------------------------------------------
// nearest-junction matching per line endpoint; emit lines_adj + keep
// ---------------------------------------------------------------------------
__global__ void __launch_bounds__(256)
match_kernel(const float* __restrict__ lines_pred, const float* __restrict__ juncs,
             float* __restrict__ lines_adj_out, float* __restrict__ lines_adj_ws,
             float* __restrict__ keep_ws) {
    __shared__ float sj[2 * N_TOPK];
    for (int i = threadIdx.x; i < 2 * N_TOPK; i += blockDim.x) sj[i] = juncs[i];
    __syncthreads();
    int l = blockIdx.x * blockDim.x + threadIdx.x;
    if (l >= N_LINES) return;
    float ax = lines_pred[l * 4 + 0], ay = lines_pred[l * 4 + 1];
    float bx = lines_pred[l * 4 + 2], by = lines_pred[l * 4 + 3];
    float bd1 = INFINITY, bd2 = INFINITY;
    int i1 = 0, i2 = 0;
    for (int j = 0; j < N_TOPK; ++j) {
        float jx = sj[2 * j], jy = sj[2 * j + 1];
        float dx1 = ax - jx, dy1 = ay - jy;
        float dx2 = bx - jx, dy2 = by - jy;
        float d1 = dx1 * dx1 + dy1 * dy1;
        float d2 = dx2 * dx2 + dy2 * dy2;
        if (d1 < bd1) { bd1 = d1; i1 = j; }
        if (d2 < bd2) { bd2 = d2; i2 = j; }
    }
    int imin = (i1 < i2) ? i1 : i2;
    int imax = (i1 < i2) ? i2 : i1;
    keep_ws[l] = (imin < imax) ? 1.0f : 0.0f;
    float o0 = sj[2 * imin], o1 = sj[2 * imin + 1];
    float o2 = sj[2 * imax], o3 = sj[2 * imax + 1];
    lines_adj_out[l * 4 + 0] = o0; lines_adj_out[l * 4 + 1] = o1;
    lines_adj_out[l * 4 + 2] = o2; lines_adj_out[l * 4 + 3] = o3;
    lines_adj_ws[l * 4 + 0] = o0; lines_adj_ws[l * 4 + 1] = o1;
    lines_adj_ws[l * 4 + 2] = o2; lines_adj_ws[l * 4 + 3] = o3;
}

// ---------------------------------------------------------------------------
// loi GEMM via WMMA: loiT[hw][d] = sum_c featuresH[c][hw] * fc1_wH[d][c] + fc1_b[d]
// M = 16384 (hw), N = 128 (d), K = 256 (c). Wave computes 16x64 strip.
// ---------------------------------------------------------------------------
__global__ void __launch_bounds__(256)
loi_gemm_kernel(const _Float16* __restrict__ FH,  // [256][16384] c-major
                const _Float16* __restrict__ WH,  // [128][256]   row-major
                const float*    __restrict__ bias,
                float*          __restrict__ loiT) {
    const int K = C_FEAT;
    int wid  = (blockIdx.x * blockDim.x + threadIdx.x) >> 5;
    int lane = threadIdx.x & 31;
    if (wid >= (HW / 16) * 2) return;
    int mt = wid >> 1;                       // 1024 m-tiles
    int n0 = (wid & 1) * 64;                 // 2 n-groups of 64
    int m0 = mt << 4;
    int row = lane & 15;
    int hi  = lane >> 4;
    int m   = m0 + row;

    v8f acc[4] = {};
    for (int k0 = 0; k0 < K; k0 += 32) {
        v16h a;
        #pragma unroll
        for (int i = 0; i < 8; ++i)
            a[i]     = FH[(size_t)(k0 + hi * 16 + i) * HW + m];
        #pragma unroll
        for (int i = 0; i < 8; ++i)
            a[8 + i] = FH[(size_t)(k0 + hi * 16 + 8 + i) * HW + m];
        #pragma unroll
        for (int t = 0; t < 4; ++t) {
            v16h b = *(const v16h*)(WH + (size_t)(n0 + t * 16 + row) * K + k0 + hi * 16);
            acc[t] = __builtin_amdgcn_wmma_f32_16x16x32_f16(
                false, a, false, b, (short)0, acc[t], false, false);
        }
    }
    #pragma unroll
    for (int t = 0; t < 4; ++t) {
        int col = n0 + t * 16 + row;
        float bv = bias[col];
        #pragma unroll
        for (int r = 0; r < 8; ++r) {
            int mr = m0 + hi * 8 + r;
            loiT[(size_t)mr * DIM_LOI + col] = acc[t][r] + bv;
        }
    }
}

// ---------------------------------------------------------------------------
// line pooling: 32 samples along each line, bilinear from loiT, max over
// groups of 4 -> feat[line][d*8+p] (f16). One block per line, thread = channel.
// ---------------------------------------------------------------------------
__global__ void __launch_bounds__(128)
pooling_kernel(const float* __restrict__ loiT, const float* __restrict__ linesAdj,
               _Float16* __restrict__ featH) {
    int l = blockIdx.x;
    int d = threadIdx.x;
    float x1 = linesAdj[l * 4 + 0], y1 = linesAdj[l * 4 + 1];
    float x2 = linesAdj[l * 4 + 2], y2 = linesAdj[l * 4 + 3];
    #pragma unroll
    for (int p = 0; p < 8; ++p) {
        float mx = -INFINITY;
        #pragma unroll
        for (int s = 0; s < 4; ++s) {
            int j = p * 4 + s;
            float t = (float)j * (1.0f / 31.0f);
            float px = x1 * t + x2 * (1.0f - t) - 0.5f;
            float py = y1 * t + y2 * (1.0f - t) - 0.5f;
            float px0 = fminf(fmaxf(floorf(px), 0.0f), 127.0f);
            float py0 = fminf(fmaxf(floorf(py), 0.0f), 127.0f);
            float px1 = fminf(px0 + 1.0f, 127.0f);
            float py1 = fminf(py0 + 1.0f, 127.0f);
            int ix0 = (int)px0, iy0 = (int)py0, ix1 = (int)px1, iy1 = (int)py1;
            float wx0 = px - px0, wx1 = px1 - px;
            float wy0 = py - py0, wy1 = py1 - py;
            float v =
                loiT[((size_t)(iy0 * IMG_W + ix0)) * DIM_LOI + d] * wy1 * wx1 +
                loiT[((size_t)(iy1 * IMG_W + ix0)) * DIM_LOI + d] * wy0 * wx1 +
                loiT[((size_t)(iy0 * IMG_W + ix1)) * DIM_LOI + d] * wy1 * wx0 +
                loiT[((size_t)(iy1 * IMG_W + ix1)) * DIM_LOI + d] * wy0 * wx0;
            mx = fmaxf(mx, v);
        }
        featH[(size_t)l * 1024 + d * 8 + p] = (_Float16)mx;
    }
}

// ---------------------------------------------------------------------------
// MLP layer via WMMA: Out = act(A @ W^T + bias), A: MxK f16, W: NxK f16.
// Wave computes 16x64 strip (1 A fragment reused over 4 B fragments).
// ---------------------------------------------------------------------------
__global__ void __launch_bounds__(256)
mlp_gemm_kernel(const _Float16* __restrict__ A, const _Float16* __restrict__ W,
                const float* __restrict__ bias, _Float16* __restrict__ Out,
                int M, int N, int K, int relu) {
    int wid  = (blockIdx.x * blockDim.x + threadIdx.x) >> 5;
    int lane = threadIdx.x & 31;
    int nGroups = N >> 6;
    int totalWaves = (M >> 4) * nGroups;
    if (wid >= totalWaves) return;
    int mt = wid / nGroups;
    int ng = wid % nGroups;
    int m0 = mt << 4;
    int n0 = ng << 6;
    int row = lane & 15;
    int hi  = lane >> 4;

    const _Float16* arow = A + (size_t)(m0 + row) * K + hi * 16;
    v8f acc[4] = {};
    for (int k0 = 0; k0 < K; k0 += 32) {
        union { v16h v; v8h h[2]; } ua;
        ua.h[0] = *(const v8h*)(arow + k0);
        ua.h[1] = *(const v8h*)(arow + k0 + 8);
        #pragma unroll
        for (int t = 0; t < 4; ++t) {
            v16h b = *(const v16h*)(W + (size_t)(n0 + t * 16 + row) * K + k0 + hi * 16);
            acc[t] = __builtin_amdgcn_wmma_f32_16x16x32_f16(
                false, ua.v, false, b, (short)0, acc[t], false, false);
        }
    }
    #pragma unroll
    for (int t = 0; t < 4; ++t) {
        int col = n0 + t * 16 + row;
        float bv = bias[col];
        #pragma unroll
        for (int r = 0; r < 8; ++r) {
            int mr = m0 + hi * 8 + r;
            float v = acc[t][r] + bv;
            if (relu) v = fmaxf(v, 0.0f);
            Out[(size_t)mr * N + col] = (_Float16)v;
        }
    }
}

// ---------------------------------------------------------------------------
// final layer: logits = hid2 @ w3 + b3; scores = sigmoid(logits) * keep
// ---------------------------------------------------------------------------
__global__ void __launch_bounds__(128)
final_kernel(const _Float16* __restrict__ hid2, const float* __restrict__ w3,
             const float* __restrict__ b3, const float* __restrict__ keep,
             float* __restrict__ scores) {
    __shared__ float red[128];
    int l = blockIdx.x, t = threadIdx.x;
    float s = 0.0f;
    for (int k = t; k < DIM_FC; k += 128)
        s += (float)hid2[(size_t)l * DIM_FC + k] * w3[k];
    red[t] = s;
    __syncthreads();
    for (int st = 64; st > 0; st >>= 1) {
        if (t < st) red[t] += red[t + st];
        __syncthreads();
    }
    if (t == 0) {
        float logit = red[0] + b3[0];
        scores[l] = (1.0f / (1.0f + expf(-logit))) * keep[l];
    }
}

// ---------------------------------------------------------------------------
extern "C" void kernel_launch(void* const* d_in, const int* in_sizes, int n_in,
                              void* d_out, int out_size, void* d_ws, size_t ws_size,
                              hipStream_t stream) {
    const float* out9       = (const float*)d_in[0];
    const float* features   = (const float*)d_in[1];
    const float* lines_pred = (const float*)d_in[2];
    const float* fc1_w      = (const float*)d_in[3];
    const float* fc1_b      = (const float*)d_in[4];
    const float* w1         = (const float*)d_in[5];
    const float* b1         = (const float*)d_in[6];
    const float* w2         = (const float*)d_in[7];
    const float* b2         = (const float*)d_in[8];
    const float* w3         = (const float*)d_in[9];
    const float* b3         = (const float*)d_in[10];

    float* scores_out   = (float*)d_out;               // 10000
    float* linesadj_out = scores_out + N_LINES;        // 40000
    float* juncs_out    = scores_out + 50000;          // 600
    float* jscores_out  = scores_out + 50600;          // 300

    char* cur = (char*)d_ws;
    auto alloc = [&](size_t bytes) -> void* {
        void* r = (void*)cur;
        cur += (bytes + 255) & ~(size_t)255;
        return r;
    };
    float*    jflat       = (float*)alloc((size_t)HW * 4);
    float*    joff        = (float*)alloc((size_t)2 * HW * 4);
    float*    juncs_ws    = (float*)alloc(2 * N_TOPK * 4);
    float*    keep_ws     = (float*)alloc(N_LINES * 4);
    float*    linesadj_ws = (float*)alloc((size_t)N_LINES * 4 * 4);
    _Float16* featuresH   = (_Float16*)alloc((size_t)C_FEAT * HW * 2);
    _Float16* fc1_wH      = (_Float16*)alloc((size_t)DIM_LOI * C_FEAT * 2);
    _Float16* w1H         = (_Float16*)alloc((size_t)DIM_FC * DIM_FC * 2);
    _Float16* w2H         = (_Float16*)alloc((size_t)DIM_FC * DIM_FC * 2);
    float*    loiT        = (float*)alloc((size_t)HW * DIM_LOI * 4);
    _Float16* featH       = (_Float16*)alloc((size_t)N_LINES * DIM_FC * 2);
    _Float16* hid1H       = (_Float16*)alloc((size_t)N_LINES * DIM_FC * 2);
    _Float16* hid2H       = featH;   // reuse: layer2 reads hid1H, feat no longer needed

    // f32 -> f16 conversions
    {
        int n = C_FEAT * HW;
        convert_f16_kernel<<<(n + 255) / 256, 256, 0, stream>>>(features, featuresH, n);
        n = DIM_LOI * C_FEAT;
        convert_f16_kernel<<<(n + 255) / 256, 256, 0, stream>>>(fc1_w, fc1_wH, n);
        n = DIM_FC * DIM_FC;
        convert_f16_kernel<<<(n + 255) / 256, 256, 0, stream>>>(w1, w1H, n);
        convert_f16_kernel<<<(n + 255) / 256, 256, 0, stream>>>(w2, w2H, n);
    }

    // junctions: softmax + NMS + offsets
    junction_prep_kernel<<<HW / 256, 256, 0, stream>>>(out9, jflat, joff);

    // top-300 + junction coordinates
    topk_kernel<<<1, 1024, 0, stream>>>(jflat, joff, jscores_out, juncs_out, juncs_ws);

    // line-endpoint matching
    match_kernel<<<(N_LINES + 255) / 256, 256, 0, stream>>>(
        lines_pred, juncs_ws, linesadj_out, linesadj_ws, keep_ws);

    // loi GEMM (WMMA): 2048 waves
    loi_gemm_kernel<<<(2048 * 32) / 256, 256, 0, stream>>>(featuresH, fc1_wH, fc1_b, loiT);

    // line pooling -> f16 features
    pooling_kernel<<<N_LINES, 128, 0, stream>>>(loiT, linesadj_ws, featH);

    // MLP layers (WMMA): 625*16 = 10000 waves each
    mlp_gemm_kernel<<<(10000 * 32) / 256, 256, 0, stream>>>(
        featH, w1H, b1, hid1H, N_LINES, DIM_FC, DIM_FC, 1);
    mlp_gemm_kernel<<<(10000 * 32) / 256, 256, 0, stream>>>(
        hid1H, w2H, b2, hid2H, N_LINES, DIM_FC, DIM_FC, 1);

    // final scores
    final_kernel<<<N_LINES, 128, 0, stream>>>(hid2H, w3, b3, keep_ws, scores_out);
}
